// Model_18640158065043
// MI455X (gfx1250) — compile-verified, matrix operand order checked by reading, not run on hardware
//
#include <hip/hip_runtime.h>

// ---------------------------------------------------------------------------
// GRIP-style ST-GCN + seq2seq GRU for MI455X (gfx1250), wave32 + WMMA f16.
// All GEMMs use v_wmma_f32_16x16x32_f16 with f32 accumulation; BN/ReLU/bias/
// residual fused into WMMA epilogues; GRU scan is a persistent per-block loop.
// B operands are staged TRANSPOSED in LDS so every WMMA fragment load is
// 2 x ds_load_b128 per lane; K-strides padded (72/136 halves) to spread banks.
// ---------------------------------------------------------------------------

typedef __attribute__((ext_vector_type(16))) _Float16 v16h;
typedef __attribute__((ext_vector_type(8)))  _Float16 h8;
typedef __attribute__((ext_vector_type(8)))  float    v8f;

#define DEV static __device__ __forceinline__

constexpr int kN  = 64;          // batch
constexpr int kT  = 64;          // time
constexpr int kV  = 120;         // vertices
constexpr int kTV = kT * kV;     // 7680
constexpr int kB  = kN * kV;     // 7680 GRU rows
constexpr int kH  = 60;          // GRU hidden
constexpr int kG  = 180;         // 3*H gates
constexpr int LD64  = 72;        // padded K-stride for K<=64 tiles (16B-mult)
constexpr int LD128 = 136;       // padded K-stride for K<=128 tiles
constexpr float kEPS = 1e-5f;

// ---- WMMA fragment load (wave32 layout per CDNA5 ISA 7.12.2) --------------
// Works for A (row-major [M][ldk]) and for B staged transposed ([N][ldk]).
// Per lane: row = lane&15, k0 = (lane&16)?8:0 ; values at k0+0..7, 16+k0+0..7
// -> two contiguous 16B chunks -> 2x ds_load_b128.
DEV v16h frag_ld(const _Float16* p, int ldk, int lane) {
  const _Float16* q = p + (lane & 15) * ldk + ((lane & 16) ? 8 : 0);
  h8 lo = *(const h8*)(q);
  h8 hi = *(const h8*)(q + 16);
  return __builtin_shufflevector(lo, hi, 0, 1, 2, 3, 4, 5, 6, 7,
                                 8, 9, 10, 11, 12, 13, 14, 15);
}
DEV v8f wmma16(v16h a, v16h b, v8f c) {
  return __builtin_amdgcn_wmma_f32_16x16x32_f16(false, a, false, b, (short)0, c,
                                                false, false);
}
DEV float sigmf(float x) { return 1.0f / (1.0f + __expf(-x)); }

// ---------------------------------------------------------------------------
// 1) BatchNorm on input [N,4,T,V]
__global__ void k_bn0(const float* __restrict__ x, const float* g, const float* b,
                      const float* m, const float* v, float* __restrict__ o,
                      int C, int total) {
  int i = blockIdx.x * blockDim.x + threadIdx.x;
  if (i >= total) return;
  int c = (i / kTV) % C;
  float inv = g[c] * rsqrtf(v[c] + kEPS);
  o[i] = (x[i] - m[c]) * inv + b[c];
}

// 2) 1x1 conv with C=4 (tiny K: VALU). res path (f32 + BN) and gcn0 path (f16)
__global__ void k_conv4_bn(const float* __restrict__ x, const float* W,
                           const float* bias, const float* bg, const float* bb,
                           const float* bm, const float* bv,
                           float* __restrict__ out, int O) {
  int i = blockIdx.x * blockDim.x + threadIdx.x;
  int total = kN * O * kTV;
  if (i >= total) return;
  int p = i % kTV, o = (i / kTV) % O, n = i / (kTV * O);
  const float* xb = x + (size_t)n * 4 * kTV + p;
  float acc = bias[o];
#pragma unroll
  for (int c = 0; c < 4; ++c) acc += W[o * 4 + c] * xb[(size_t)c * kTV];
  float inv = bg[o] * rsqrtf(bv[o] + kEPS);
  out[i] = (acc - bm[o]) * inv + bb[o];
}
__global__ void k_conv4_h(const float* __restrict__ x, const float* W,
                          const float* bias, _Float16* __restrict__ out, int O) {
  int i = blockIdx.x * blockDim.x + threadIdx.x;
  int total = kN * O * kTV;
  if (i >= total) return;
  int p = i % kTV, o = (i / kTV) % O, n = i / (kTV * O);
  const float* xb = x + (size_t)n * 4 * kTV + p;
  float acc = bias[o];
#pragma unroll
  for (int c = 0; c < 4; ++c) acc += W[o * 4 + c] * xb[(size_t)c * kTV];
  out[i] = (_Float16)acc;
}

// 3) GCN 1x1 conv, C=64 -> O=128, WMMA. One block per (n,t). 8 waves.
__global__ void k_gcn(const _Float16* __restrict__ x16, const float* __restrict__ W,
                      const float* bias, _Float16* __restrict__ y16) {
  __shared__ __align__(16) _Float16 sW[128 * LD64];   // A: [o][c]
  __shared__ __align__(16) _Float16 sXT[128 * LD64];  // B^T: [v][c]
  int n = blockIdx.x / kT, t = blockIdx.x % kT;
  int tid = threadIdx.x, lane = tid & 31, wv = tid >> 5;
  for (int i = tid; i < 128 * 64; i += 256) {
    int o = i >> 6, c = i & 63;
    sW[o * LD64 + c] = (_Float16)W[i];
  }
  for (int i = tid; i < 128 * 64; i += 256) {
    int vx = i & 127, c = i >> 7;
    sXT[vx * LD64 + c] =
        (vx < kV) ? x16[(((size_t)n * 64 + c) * kT + t) * kV + vx] : (_Float16)0.f;
  }
  __syncthreads();
  for (int nt = 0; nt < 8; ++nt) {           // wave wv owns Mtile wv (of 8)
    v8f acc = {};
#pragma unroll
    for (int s = 0; s < 2; ++s)
      acc = wmma16(frag_ld(sW + wv * 16 * LD64 + s * 32, LD64, lane),
                   frag_ld(sXT + nt * 16 * LD64 + s * 32, LD64, lane), acc);
    int col = (lane & 15) + nt * 16;
    if (col < kV) {
      int rbase = wv * 16 + ((lane & 16) ? 8 : 0);
#pragma unroll
      for (int q = 0; q < 8; ++q) {
        int o = rbase + q;
        y16[(((size_t)n * 128 + o) * kT + t) * kV + col] =
            (_Float16)(acc[q] + bias[o]);
      }
    }
  }
}

// 4) Graph message passing: out[c,w] = sum_k sum_v y[k,c,v] * (fixed+imp)[k,v,w]
//    fused with BN1 + ReLU epilogue -> f16. Block per (n, 4 timesteps).
__global__ void k_msgpass(const _Float16* __restrict__ y16,
                          const float* __restrict__ fixed, const float* __restrict__ imp,
                          const float* bg, const float* bb, const float* bm,
                          const float* bv, _Float16* __restrict__ h16) {
  __shared__ __align__(16) _Float16 sAT[2][128 * LD128];  // B^T: [w][v] per k
  __shared__ __align__(16) _Float16 sY[2][64 * LD128];    // A: [c][v] per k
  int n = blockIdx.x / (kT / 4), t0 = (blockIdx.x % (kT / 4)) * 4;
  int tid = threadIdx.x, lane = tid & 31, wv = tid >> 5;
  for (int i = tid; i < 2 * 128 * 128; i += 256) {
    int k = i >> 14, r = i & 16383, w = r & 127, vx = r >> 7;
    float val = 0.f;
    if (vx < kV && w < kV)
      val = fixed[(((size_t)n * 2 + k) * kV + vx) * kV + w] +
            imp[((size_t)k * kV + vx) * kV + w];
    sAT[k][w * LD128 + vx] = (_Float16)val;
  }
  __syncthreads();
  for (int tt = 0; tt < 4; ++tt) {
    int t = t0 + tt;
    for (int i = tid; i < 2 * 64 * 128; i += 256) {
      int k = i >> 13, r = i & 8191, vx = r & 127, c = r >> 7;
      sY[k][c * LD128 + vx] =
          (vx < kV) ? y16[(((size_t)n * 128 + k * 64 + c) * kT + t) * kV + vx]
                    : (_Float16)0.f;
    }
    __syncthreads();
    for (int mt = 0; mt < 4; ++mt) {        // wave wv owns Ntile wv (of 8)
      v8f acc = {};
#pragma unroll
      for (int k = 0; k < 2; ++k)
#pragma unroll
        for (int s = 0; s < 4; ++s)
          acc = wmma16(frag_ld(sY[k] + mt * 16 * LD128 + s * 32, LD128, lane),
                       frag_ld(sAT[k] + wv * 16 * LD128 + s * 32, LD128, lane),
                       acc);
      int w = (lane & 15) + wv * 16;
      if (w < kV) {
        int rbase = mt * 16 + ((lane & 16) ? 8 : 0);
#pragma unroll
        for (int q = 0; q < 8; ++q) {
          int c = rbase + q;
          float inv = bg[c] * rsqrtf(bv[c] + kEPS);
          float val = (acc[q] - bm[c]) * inv + bb[c];
          h16[(((size_t)n * 64 + c) * kT + t) * kV + w] =
              (_Float16)fmaxf(val, 0.f);
        }
      }
    }
    __syncthreads();
  }
}

// 5) Temporal conv 5x1 over T + bias + BN2 + residual + ReLU -> f32 and f16.
//    Block per (n, 2 timesteps). 5 taps = 5 accumulated K=64 sub-GEMMs.
__global__ void k_tconv(const _Float16* __restrict__ h16, const float* __restrict__ tW,
                        const float* tb, const float* bg, const float* bb,
                        const float* bm, const float* bv,
                        const float* __restrict__ res, float* __restrict__ xo,
                        _Float16* __restrict__ xo16) {
  __shared__ __align__(16) _Float16 sXT[6][128 * LD64];  // B^T: [v][c] slices
  __shared__ __align__(16) _Float16 sW[5][64 * LD64];    // A: [o][i] per tap
  int n = blockIdx.x / (kT / 2), t0 = (blockIdx.x % (kT / 2)) * 2;
  int tid = threadIdx.x, lane = tid & 31, wv = tid >> 5;
  for (int i = tid; i < 5 * 64 * 64; i += 256) {
    int d = i / 4096, r = i % 4096, o = r >> 6, ii = r & 63;
    sW[d][o * LD64 + ii] = (_Float16)tW[((size_t)o * 64 + ii) * 5 + d];
  }
  for (int i = tid; i < 6 * 128 * 64; i += 256) {
    int sl = i >> 13, r = i & 8191, vx = r & 127, c = r >> 7;
    int t = t0 - 2 + sl;
    _Float16 val = (_Float16)0.f;
    if (vx < kV && t >= 0 && t < kT)
      val = h16[(((size_t)n * 64 + c) * kT + t) * kV + vx];
    sXT[sl][vx * LD64 + c] = val;
  }
  __syncthreads();
  for (int tt = 0; tt < 2; ++tt) {
    int t = t0 + tt;
    for (int mt = 0; mt < 4; ++mt) {        // wave wv owns Ntile wv
      v8f acc = {};
      for (int d = 0; d < 5; ++d)
#pragma unroll
        for (int s = 0; s < 2; ++s)
          acc = wmma16(frag_ld(sW[d] + mt * 16 * LD64 + s * 32, LD64, lane),
                       frag_ld(sXT[tt + d] + wv * 16 * LD64 + s * 32, LD64, lane),
                       acc);
      int vx = (lane & 15) + wv * 16;
      if (vx < kV) {
        int rbase = mt * 16 + ((lane & 16) ? 8 : 0);
#pragma unroll
        for (int q = 0; q < 8; ++q) {
          int o = rbase + q;
          float val = acc[q] + tb[o];
          float inv = bg[o] * rsqrtf(bv[o] + kEPS);
          val = (val - bm[o]) * inv + bb[o];
          size_t idx = (((size_t)n * 64 + o) * kT + t) * kV + vx;
          val = fmaxf(val + res[idx], 0.f);
          xo[idx] = val;
          xo16[idx] = (_Float16)val;
        }
      }
    }
  }
}

// 6) feat[(n*V+v), t, c] = x[n,c,t,v] (f16) ; last_loc from raw input
__global__ void k_feat(const float* __restrict__ x, _Float16* __restrict__ feat) {
  int i = blockIdx.x * blockDim.x + threadIdx.x;
  int total = kN * 64 * kT * kV;
  if (i >= total) return;
  int c = i & 63, t = (i >> 6) % kT, b = i >> 12;
  int n = b / kV, v = b % kV;
  feat[i] = (_Float16)x[(((size_t)n * 64 + c) * kT + t) * kV + v];
}
__global__ void k_last(const float* __restrict__ af, float* __restrict__ last) {
  int b = blockIdx.x * blockDim.x + threadIdx.x;
  if (b >= kB) return;
  int n = b / kV, v = b % kV;
  last[b * 2 + 0] = af[(((size_t)n * 4 + 0) * kT + (kT - 1)) * kV + v];
  last[b * 2 + 1] = af[(((size_t)n * 4 + 1) * kT + (kT - 1)) * kV + v];
}

// 7) Input projection GEMM: out[M,180] = in16[M,K] @ W^T + bias. 128 rows/block.
//    W is [180][K] row-major == already B^T: direct coalesced copy into LDS.
__global__ void k_xp(const _Float16* __restrict__ in16, int ldin, int K,
                     const float* __restrict__ W, const float* bias,
                     float* __restrict__ out) {
  __shared__ __align__(16) _Float16 sA[128 * LD64];
  __shared__ __align__(16) _Float16 sBT[192 * LD64];
  int r0 = blockIdx.x * 128;
  int tid = threadIdx.x, lane = tid & 31, wv = tid >> 5;
  for (int i = tid; i < 128 * 64; i += 256) {
    int r = i >> 6, k = i & 63;
    sA[r * LD64 + k] = (k < K) ? in16[(size_t)(r0 + r) * ldin + k] : (_Float16)0.f;
  }
  for (int i = tid; i < 192 * 64; i += 256) {
    int g = i >> 6, k = i & 63;
    sBT[g * LD64 + k] =
        (k < K && g < kG) ? (_Float16)W[(size_t)g * K + k] : (_Float16)0.f;
  }
  __syncthreads();
  for (int nt = 0; nt < 12; ++nt) {          // wave wv owns Mtile wv (of 8)
    v8f acc = {};
#pragma unroll
    for (int s = 0; s < 2; ++s)
      acc = wmma16(frag_ld(sA + wv * 16 * LD64 + s * 32, LD64, lane),
                   frag_ld(sBT + nt * 16 * LD64 + s * 32, LD64, lane), acc);
    int g = (lane & 15) + nt * 16;
    if (g < kG) {
      int rr = wv * 16 + ((lane & 16) ? 8 : 0);
#pragma unroll
      for (int q = 0; q < 8; ++q)
        out[(size_t)(r0 + rr + q) * kG + g] = acc[q] + bias[g];
    }
  }
}

// 8) Persistent GRU scan: block owns 64 rows, loops all T steps.
//    gh = h @ Whh^T via WMMA each step; gates in VALU; h kept in LDS (f32+f16).
__global__ void k_scan(const float* __restrict__ xp, const float* __restrict__ Whh,
                       const float* __restrict__ bhh, _Float16* __restrict__ outs,
                       float* __restrict__ hT, int writeOuts) {
  __shared__ __align__(16) _Float16 sBT[192 * LD64];
  __shared__ __align__(16) _Float16 sH16[64 * LD64];
  __shared__ float sGH[64 * kG];
  __shared__ float sHf[64 * kH];
  int r0 = blockIdx.x * 64;
  int tid = threadIdx.x, lane = tid & 31, wv = tid >> 5;
  for (int i = tid; i < 192 * 64; i += 256) {
    int g = i >> 6, k = i & 63;
    sBT[g * LD64 + k] =
        (k < kH && g < kG) ? (_Float16)Whh[(size_t)g * kH + k] : (_Float16)0.f;
  }
  for (int i = tid; i < 64 * LD64; i += 256) sH16[i] = (_Float16)0.f;
  for (int i = tid; i < 64 * kH; i += 256) sHf[i] = 0.f;
  __syncthreads();
  for (int t = 0; t < kT; ++t) {
    for (int j = 0; j < 6; ++j) {            // 48 tiles over 8 waves
      int flat = wv * 6 + j, mt = flat / 12, nt = flat % 12;
      v8f acc = {};
#pragma unroll
      for (int s = 0; s < 2; ++s)
        acc = wmma16(frag_ld(sH16 + mt * 16 * LD64 + s * 32, LD64, lane),
                     frag_ld(sBT + nt * 16 * LD64 + s * 32, LD64, lane), acc);
      int g = (lane & 15) + nt * 16;
      if (g < kG) {
        int rr = mt * 16 + ((lane & 16) ? 8 : 0);
#pragma unroll
        for (int q = 0; q < 8; ++q) sGH[(rr + q) * kG + g] = acc[q];
      }
    }
    __syncthreads();
    for (int e = tid; e < 64 * kH; e += 256) {
      int r = e / kH, j = e % kH;
      const float* xr = xp + ((size_t)(r0 + r) * kT + t) * kG;
      if (t + 1 < kT) __builtin_prefetch(xr + kG, 0, 0);
      float ir = xr[j], iz = xr[kH + j], inn = xr[2 * kH + j];
      float hr = sGH[r * kG + j] + bhh[j];
      float hz = sGH[r * kG + kH + j] + bhh[kH + j];
      float hn = sGH[r * kG + 2 * kH + j] + bhh[2 * kH + j];
      float rg = sigmf(ir + hr), zg = sigmf(iz + hz);
      float ng = tanhf(inn + rg * hn);
      float h2 = (1.f - zg) * ng + zg * sHf[e];
      sHf[e] = h2;
      sH16[r * LD64 + j] = (_Float16)h2;
      if (writeOuts)
        outs[((size_t)(r0 + r) * kT + t) * kH + j] = (_Float16)h2;
    }
    __syncthreads();
  }
  for (int e = tid; e < 64 * kH; e += 256)
    hT[(size_t)(r0 + e / kH) * kH + (e % kH)] = sHf[e];
}

// ---- decoder helpers: 32x180 GEMM from LDS f16 operands -------------------
DEV void gemm32x180(const _Float16* sA, const _Float16* sBT, float* sOut,
                    int lane, int wv) {
  for (int j = 0; j < 3; ++j) {              // 24 tiles over 8 waves
    int flat = wv * 3 + j, mt = flat / 12, nt = flat % 12;
    v8f acc = {};
#pragma unroll
    for (int s = 0; s < 2; ++s)
      acc = wmma16(frag_ld(sA + mt * 16 * LD64 + s * 32, LD64, lane),
                   frag_ld(sBT + nt * 16 * LD64 + s * 32, LD64, lane), acc);
    int g = (lane & 15) + nt * 16;
    if (g < kG) {
      int rr = mt * 16 + ((lane & 16) ? 8 : 0);
#pragma unroll
      for (int q = 0; q < 8; ++q) sOut[(rr + q) * kG + g] = acc[q];
    }
  }
}
DEV void stage_w(const float* W, int K, _Float16* sBT, int tid) {
  for (int i = tid; i < 192 * 64; i += 256) {
    int g = i >> 6, k = i & 63;
    sBT[g * LD64 + k] =
        (k < K && g < kG) ? (_Float16)W[(size_t)g * K + k] : (_Float16)0.f;
  }
}

// 9) Decoder: hidden is reset to encoder state every step, so gh0/gh1 are
//    precomputed once (WMMA); the 12-step loop only does K=2 / K=60 work.
__global__ void k_dec(const float* __restrict__ h0T, const float* __restrict__ h1T,
                      const float* __restrict__ last,
                      const float* Whh0, const float* Wih0, const float* bhh0,
                      const float* bih0, const float* Whh1, const float* Wih1,
                      const float* bhh1, const float* bih1,
                      const float* linW, const float* linb,
                      const int* __restrict__ plen, float* __restrict__ pred,
                      int first) {
  __shared__ __align__(16) _Float16 sBT[192 * LD64];
  __shared__ __align__(16) _Float16 sA16[32 * LD64];
  __shared__ float sGH0[32 * kG];
  __shared__ float sGH1[32 * kG];
  __shared__ float sGI[32 * kG];
  __shared__ float sH0[32 * kH];
  __shared__ float sH1[32 * kH];
  __shared__ float sBf[32 * kH];
  __shared__ float sInp[32 * 2];
  int r0 = blockIdx.x * 32;
  int tid = threadIdx.x, lane = tid & 31, wv = tid >> 5;
  int L = plen[0]; if (L > 12) L = 12; if (L < 0) L = 0;

  for (int i = tid; i < 32 * kH; i += 256) {
    int r = i / kH, k = i % kH;
    sH0[i] = h0T[(size_t)(r0 + r) * kH + k];
    sH1[i] = h1T[(size_t)(r0 + r) * kH + k];
  }
  for (int i = tid; i < 32 * 64; i += 256) {
    int r = i >> 6, k = i & 63;
    sA16[r * LD64 + k] =
        (k < kH) ? (_Float16)h0T[(size_t)(r0 + r) * kH + k] : (_Float16)0.f;
  }
  for (int i = tid; i < 64; i += 256)
    sInp[i] = last[(size_t)(r0 + (i >> 1)) * 2 + (i & 1)];
  stage_w(Whh0, kH, sBT, tid);
  __syncthreads();
  gemm32x180(sA16, sBT, sGH0, lane, wv);     // gh0 = h0 @ Whh0^T (bias in gates)
  __syncthreads();
  for (int i = tid; i < 32 * 64; i += 256) {
    int r = i >> 6, k = i & 63;
    sA16[r * LD64 + k] =
        (k < kH) ? (_Float16)h1T[(size_t)(r0 + r) * kH + k] : (_Float16)0.f;
  }
  stage_w(Whh1, kH, sBT, tid);
  __syncthreads();
  gemm32x180(sA16, sBT, sGH1, lane, wv);     // gh1 = h1 @ Whh1^T
  __syncthreads();
  stage_w(Wih1, kH, sBT, tid);               // keep Wih1 resident for the loop
  __syncthreads();

  for (int l = 0; l < L; ++l) {
    // layer0 cell: K=2 input projection in VALU, gates vs precomputed gh0
    for (int e = tid; e < 32 * kH; e += 256) {
      int r = e / kH, j = e % kH;
      float x0 = sInp[r * 2], x1 = sInp[r * 2 + 1];
      float gir = bih0[j] + Wih0[j * 2] * x0 + Wih0[j * 2 + 1] * x1;
      float giz = bih0[kH + j] + Wih0[(kH + j) * 2] * x0 + Wih0[(kH + j) * 2 + 1] * x1;
      float gin = bih0[2 * kH + j] + Wih0[(2 * kH + j) * 2] * x0 + Wih0[(2 * kH + j) * 2 + 1] * x1;
      float hr = sGH0[r * kG + j] + bhh0[j];
      float hz = sGH0[r * kG + kH + j] + bhh0[kH + j];
      float hn = sGH0[r * kG + 2 * kH + j] + bhh0[2 * kH + j];
      float rg = sigmf(gir + hr), zg = sigmf(giz + hz);
      float ng = tanhf(gin + rg * hn);
      sA16[r * LD64 + j] = (_Float16)((1.f - zg) * ng + zg * sH0[e]);
    }
    __syncthreads();
    gemm32x180(sA16, sBT, sGI, lane, wv);    // gi1 = a @ Wih1^T
    __syncthreads();
    for (int e = tid; e < 32 * kH; e += 256) {
      int r = e / kH, j = e % kH;
      float gir = sGI[r * kG + j] + bih1[j];
      float giz = sGI[r * kG + kH + j] + bih1[kH + j];
      float gin = sGI[r * kG + 2 * kH + j] + bih1[2 * kH + j];
      float hr = sGH1[r * kG + j] + bhh1[j];
      float hz = sGH1[r * kG + kH + j] + bhh1[kH + j];
      float hn = sGH1[r * kG + 2 * kH + j] + bhh1[2 * kH + j];
      float rg = sigmf(gir + hr), zg = sigmf(giz + hz);
      float ng = tanhf(gin + rg * hn);
      sBf[e] = (1.f - zg) * ng + zg * sH1[e];
    }
    __syncthreads();
    for (int e = tid; e < 64; e += 256) {
      int r = e >> 1, c = e & 1;
      float acc = linb[c];
      for (int j = 0; j < kH; ++j) acc += linW[c * kH + j] * sBf[r * kH + j];
      acc += sInp[e];
      size_t pi = ((size_t)(r0 + r) * 12 + l) * 2 + c;
      if (first) pred[pi] = acc; else pred[pi] += acc;
      sInp[e] = acc;
    }
    __syncthreads();
  }
}

// 10) average 3 seq2seq heads and reshape [B,L,2] -> [N,2,L,V]
__global__ void k_out(const float* __restrict__ pred, float* __restrict__ out) {
  int i = blockIdx.x * blockDim.x + threadIdx.x;
  int total = kN * 2 * 12 * kV;
  if (i >= total) return;
  int v = i % kV, l = (i / kV) % 12, c = (i / (kV * 12)) % 2, n = i / (kV * 12 * 2);
  out[i] = pred[(((size_t)n * kV + v) * 12 + l) * 2 + c] * (1.f / 3.f);
}

// ---------------------------------------------------------------------------
// Workspace layout (~756 MB). xp GEMM buffer aliases the dead GCN buffers.
constexpr size_t SZX32  = (size_t)kN * 64 * kT * kV * 4;   // 125,829,120 B
constexpr size_t O_X32A = 0;
constexpr size_t O_X32B = SZX32;
constexpr size_t O_RES  = 2 * SZX32;            // later aliased by xp (354 MB fits in 377 MB)
constexpr size_t O_Y16  = O_RES + SZX32;        // [N,128,T,V] f16 == SZX32 bytes
constexpr size_t O_H16  = O_Y16 + SZX32;
constexpr size_t O_X16  = O_H16 + SZX32 / 2;
constexpr size_t O_FEAT = O_X16 + SZX32 / 2;
constexpr size_t O_OUTS = O_FEAT + SZX32 / 2;
constexpr size_t O_LAST = O_OUTS + (size_t)kB * kT * kH * 2;
constexpr size_t O_H0T  = O_LAST + (size_t)kB * 2 * 4;
constexpr size_t O_H1T  = O_H0T + (size_t)kB * kH * 4;
constexpr size_t O_PRED = O_H1T + (size_t)kB * kH * 4;

extern "C" void kernel_launch(void* const* d_in, const int* in_sizes, int n_in,
                              void* d_out, int out_size, void* d_ws, size_t ws_size,
                              hipStream_t stream) {
  (void)in_sizes; (void)out_size; (void)ws_size;
  auto F = [&](int i) { return (const float*)d_in[i]; };
  char* ws = (char*)d_ws;
  float*     x32a = (float*)(ws + O_X32A);
  float*     x32b = (float*)(ws + O_X32B);
  float*     res32 = (float*)(ws + O_RES);
  _Float16*  y16  = (_Float16*)(ws + O_Y16);
  _Float16*  h16  = (_Float16*)(ws + O_H16);
  _Float16*  x16  = (_Float16*)(ws + O_X16);
  float*     xp   = (float*)(ws + O_RES);       // alias over res32/y16/h16/x16
  _Float16*  feat = (_Float16*)(ws + O_FEAT);
  _Float16*  outs = (_Float16*)(ws + O_OUTS);
  float*     last = (float*)(ws + O_LAST);
  float*     h0T  = (float*)(ws + O_H0T);
  float*     h1T  = (float*)(ws + O_H1T);
  float*     pred = (float*)(ws + O_PRED);

  // Input index map (jax pytree / sorted-dict flattening):
  // 0 all_feature, 1 fixed_graph; params: blocks[0] 2..19, blocks[1] 20..31,
  // blocks[2] 32..43, bn0 44..47 (b,g,m,v), imp 48..50, s2s base 51+18*s
  // (dec0 Whh,Wih,bhh,bih; dec1 ...; enc0 ...; enc1 ...; lin_W, lin_b);
  // last input = predicted_length.
  const int* plen = (const int*)d_in[n_in - 1];

  // BN0
  { int total = kN * 4 * kTV;
    k_bn0<<<(total + 255) / 256, 256, 0, stream>>>(F(0), F(45), F(44), F(46), F(47), x32a, 4, total); }

  // Block 0 (C=4): residual conv+BN, GCN conv, message pass, temporal conv
  k_conv4_bn<<<(kN * 64 * kTV + 255) / 256, 256, 0, stream>>>(
      x32a, F(12), F(13), F(15), F(14), F(16), F(17), res32, 64);
  k_conv4_h<<<(kN * 128 * kTV + 255) / 256, 256, 0, stream>>>(x32a, F(10), F(11), y16, 128);
  k_msgpass<<<kN * (kT / 4), 256, 0, stream>>>(y16, F(1), F(48), F(3), F(2), F(4), F(5), h16);
  k_tconv<<<kN * (kT / 2), 256, 0, stream>>>(h16, F(18), F(19), F(7), F(6), F(8), F(9),
                                             res32, x32b, x16);
  // Blocks 1 & 2 (C=64): WMMA GCN, message pass, temporal conv (+identity res)
  for (int j = 1; j <= 2; ++j) {
    int bb = 20 + 12 * (j - 1);
    const float* xin32 = (j == 1) ? x32b : x32a;
    float* xout32 = (j == 1) ? x32a : x32b;
    k_gcn<<<kN * kT, 256, 0, stream>>>(x16, F(bb + 8), F(bb + 9), y16);
    k_msgpass<<<kN * (kT / 4), 256, 0, stream>>>(y16, F(1), F(48 + j),
                                                 F(bb + 1), F(bb + 0), F(bb + 2), F(bb + 3), h16);
    k_tconv<<<kN * (kT / 2), 256, 0, stream>>>(h16, F(bb + 10), F(bb + 11),
                                               F(bb + 5), F(bb + 4), F(bb + 6), F(bb + 7),
                                               xin32, xout32, x16);
  }

  // reshape for GRU + last locations
  k_feat<<<(kN * 64 * kTV + 255) / 256, 256, 0, stream>>>(x32b, feat);
  k_last<<<(kB + 255) / 256, 256, 0, stream>>>(F(0), last);

  // three seq2seq heads
  for (int s = 0; s < 3; ++s) {
    int p = 51 + 18 * s;
    k_xp<<<kB * kT / 128, 256, 0, stream>>>(feat, 64, 64, F(p + 9), F(p + 11), xp);
    k_scan<<<kB / 64, 256, 0, stream>>>(xp, F(p + 8), F(p + 10), outs, h0T, 1);
    k_xp<<<kB * kT / 128, 256, 0, stream>>>(outs, kH, kH, F(p + 13), F(p + 15), xp);
    k_scan<<<kB / 64, 256, 0, stream>>>(xp, F(p + 12), F(p + 14), nullptr, h1T, 0);
    k_dec<<<kB / 32, 256, 0, stream>>>(h0T, h1T, last,
                                       F(p + 0), F(p + 1), F(p + 2), F(p + 3),
                                       F(p + 4), F(p + 5), F(p + 6), F(p + 7),
                                       F(p + 16), F(p + 17), plen, pred, s == 0);
  }

  k_out<<<(kN * 2 * 12 * kV + 255) / 256, 256, 0, stream>>>(pred, (float*)d_out);
}
